// MultiHeadAttention_16947940950323
// MI455X (gfx1250) — matrix operand order repaired and checked by reading, not compiled, not run on hardware
//
#include <hip/hip_runtime.h>

// ---------------------------------------------------------------------------
// Fused MHA (QKV proj + RoPE, flash attention, output proj) for gfx1250.
// All matmuls use v_wmma_f32_16x16x32_bf16 (wave32 WMMA), f32 accumulation.
// Attention K/V^T tiles are double-buffered in LDS via CDNA5 async loads
// (global_load_async_to_lds_b128 + s_wait_asynccnt).
// ---------------------------------------------------------------------------

#define B_  4
#define S_  2048
#define H_  1024
#define NH_ 16
#define HD_ 64

typedef __attribute__((ext_vector_type(16))) __bf16 bf16x16;
typedef __attribute__((ext_vector_type(4)))  __bf16 bf16x4;
typedef __attribute__((ext_vector_type(8)))  float  v8f;
typedef __attribute__((ext_vector_type(4)))  float  f32x4;

union Frag {            // one WMMA A/B fragment: 16 bf16 per lane = 8 VGPRs
  bf16x16 v;
  f32x4   f[2];
  __bf16  e[16];
};
union F8 {              // 8 floats as 2 x 16B vectors
  f32x4 v[2];
  float s[8];
};

__device__ __forceinline__ v8f wmma_bf16(bf16x16 a, bf16x16 b, v8f c) {
  // D = A(16x32) * B(32x16) + C(16x16), f32 accum
  return __builtin_amdgcn_wmma_f32_16x16x32_bf16(false, a, false, b, (short)0, c,
                                                 false, false);
}

// CDNA5 async copy: 16B per lane, global -> LDS, tracked by ASYNCcnt.
// (inline asm: portable across ROCm7.2 / amdgpu-toolchain builtin arities)
__device__ __forceinline__ void async_load_b128(const void* gaddr, void* lds) {
  unsigned lds_off = (unsigned)(uintptr_t)lds;   // low 32 bits = LDS byte addr
  asm volatile("global_load_async_to_lds_b128 %0, %1, off"
               :: "v"(lds_off), "v"(gaddr) : "memory");
}

// ------------------------- prep kernels ------------------------------------

__global__ __launch_bounds__(256) void k_cvt(const float* __restrict__ src,
                                             __bf16* __restrict__ dst, int n4) {
  int i = blockIdx.x * 256 + threadIdx.x;
  if (i < n4) {
    f32x4 x = ((const f32x4*)src)[i];
    bf16x4 o;
    o.x = (__bf16)x.x; o.y = (__bf16)x.y; o.z = (__bf16)x.z; o.w = (__bf16)x.w;
    ((bf16x4*)dst)[i] = o;
  }
}

__global__ __launch_bounds__(256) void k_tables(float* __restrict__ cosT,
                                                float* __restrict__ sinT) {
  int i = blockIdx.x * 256 + threadIdx.x;       // 2048*32 entries
  int pos = i >> 5, j = i & 31;
  // inv_freq = 10000^(-2j/64) = 2^(-(2j/64)*log2(10000))
  float invf = exp2f(-(2.0f * (float)j / 64.0f) * 13.2877123795f);
  float ang  = (float)pos * invf;
  cosT[i] = __cosf(ang);
  sinT[i] = __sinf(ang);
}

__global__ __launch_bounds__(256) void k_bias(const int* __restrict__ mask,
                                              float* __restrict__ bias, int n) {
  int i = blockIdx.x * 256 + threadIdx.x;
  if (i < n) bias[i] = (mask[i] == 0) ? -3.0e38f : 0.0f;
}

// ------------------------- QKV projection + RoPE ---------------------------
// Computes (per 16-token tile) Q,K (RoPE'd, Q pre-scaled by 1/8) into
// [B,NH,S,HD] and V transposed into [B,NH,HD,S].  Transposed GEMM:
// A = W rows (features on WMMA-M), B = x^T (tokens on WMMA-N).

#define XPITCH (H_ + 8)   // bf16 elems; 2064B row pitch -> conflict-free LDS

__global__ __launch_bounds__(256) void k_qkv(
    const __bf16* __restrict__ xb,  const __bf16* __restrict__ Wqb,
    const __bf16* __restrict__ Wkb, const __bf16* __restrict__ Wvb,
    const float* __restrict__ cosT, const float* __restrict__ sinT,
    __bf16* __restrict__ Qr, __bf16* __restrict__ Kr, __bf16* __restrict__ Vt) {
  __shared__ __align__(16) __bf16 xs[16 * XPITCH];

  const int mbase = blockIdx.x * 16;            // global token base
  const int b     = mbase >> 11;                // / S_
  const int sbase = mbase & (S_ - 1);
  const int t     = threadIdx.x;

  // stage x tile: 16 rows x 512 bf16 (16KB)
  #pragma unroll
  for (int c = 0; c < 4; ++c) {
    int idx = t + c * 256;                      // 0..1023 chunks of 16B
    int row = idx >> 6, col = idx & 63;
    *(f32x4*)(&xs[row * XPITCH + col * 8]) =
        *(const f32x4*)(&xb[(size_t)(mbase + row) * H_ + col * 8]);
  }
  __syncthreads();

  const int wave = t >> 5, lane = t & 31;
  const int lrow = lane & 15, lhi = lane >> 4;
  const int task = blockIdx.y * 8 + wave;       // 0..47
  const int tgt  = task >> 4;                   // 0=Q 1=K 2=V
  const int head = task & 15;
  const __bf16* W = (tgt == 0) ? Wqb : ((tgt == 1) ? Wkb : Wvb);
  const int nbase = head * 64;

  v8f acc[4] = {};
  for (int k = 0; k < H_; k += 32) {
    Frag bf;                                    // B = x^T : lane = token
    const __bf16* xsrc = &xs[lrow * XPITCH + k + lhi * 8];
    bf.f[0] = *(const f32x4*)(xsrc);
    bf.f[1] = *(const f32x4*)(xsrc + 16);
    #pragma unroll
    for (int dt = 0; dt < 4; ++dt) {            // A = W rows (table-order k)
      Frag af;
      const __bf16* wsrc = &W[(size_t)(nbase + dt * 16 + lrow) * H_ + k + lhi * 8];
      af.f[0] = *(const f32x4*)(wsrc);
      af.f[1] = *(const f32x4*)(wsrc + 16);
      acc[dt] = wmma_bf16(af.v, bf.v, acc[dt]);
    }
  }

  const int s = sbase + lrow;                   // this lane's token
  const size_t bh = (size_t)b * NH_ + head;

  if (tgt < 2) {
    // RoPE: pair (d, d+32) lives at (acc[dt], acc[dt+2]) same lane.
    const float* cp = cosT + s * 32 + lhi * 8;
    const float* sp = sinT + s * 32 + lhi * 8;
    F8 c0, c1, s0, s1;
    c0.v[0] = *(const f32x4*)(cp);      c0.v[1] = *(const f32x4*)(cp + 4);
    c1.v[0] = *(const f32x4*)(cp + 16); c1.v[1] = *(const f32x4*)(cp + 20);
    s0.v[0] = *(const f32x4*)(sp);      s0.v[1] = *(const f32x4*)(sp + 4);
    s1.v[0] = *(const f32x4*)(sp + 16); s1.v[1] = *(const f32x4*)(sp + 20);
    #pragma unroll
    for (int r = 0; r < 8; ++r) {
      float x1 = acc[0][r], x2 = acc[2][r];
      acc[0][r] = x1 * c0.s[r] - x2 * s0.s[r];
      acc[2][r] = x2 * c0.s[r] + x1 * s0.s[r];
      float y1 = acc[1][r], y2 = acc[3][r];
      acc[1][r] = y1 * c1.s[r] - y2 * s1.s[r];
      acc[3][r] = y2 * c1.s[r] + y1 * s1.s[r];
    }
    const float qs = (tgt == 0) ? 0.125f : 1.0f;   // fold 1/sqrt(HD) into Q
    __bf16* dst = ((tgt == 0) ? Qr : Kr) + (bh * S_ + s) * HD_ + lhi * 8;
    #pragma unroll
    for (int dt = 0; dt < 4; ++dt) {
      Frag o;
      #pragma unroll
      for (int r = 0; r < 8; ++r) o.e[r] = (__bf16)(acc[dt][r] * qs);
      *(f32x4*)(&dst[dt * 16]) = o.f[0];
    }
  } else {
    // V stored transposed: Vt[bh, d, s]
    #pragma unroll
    for (int dt = 0; dt < 4; ++dt)
      #pragma unroll
      for (int r = 0; r < 8; ++r) {
        int d = dt * 16 + lhi * 8 + r;
        Vt[(bh * HD_ + d) * S_ + s] = (__bf16)acc[dt][r];
      }
  }
}

// ------------------------- flash attention ---------------------------------
// S^T = K Q^T (j on WMMA-M/VGPR axis, q on lane axis)  ->  softmax stats are
// per-lane + one half-swap.  O^T = V^T P^T with the P B-fragment built from
// accumulators via __shfl_xor(16) half swaps (no LDS transpose).
// K/V^T tiles stream into double-buffered LDS with async-to-LDS loads.

#define KPITCH 72   // bf16 elems (144B)
#define VPITCH 40   // bf16 elems (80B)

__global__ __launch_bounds__(256) void k_attn(
    const __bf16* __restrict__ Qr, const __bf16* __restrict__ Kr,
    const __bf16* __restrict__ Vt, const float* __restrict__ bias,
    __bf16* __restrict__ AO) {
  __shared__ __align__(16) __bf16 ks[2][32 * KPITCH];
  __shared__ __align__(16) __bf16 vs[2][64 * VPITCH];

  const int bh   = blockIdx.x;                  // 0..63
  const int b    = bh >> 4;
  const int head = bh & 15;
  const int t = threadIdx.x, wave = t >> 5, lane = t & 31;
  const int lrow = lane & 15, lhi = lane >> 4;
  const int qbase = blockIdx.y * 128 + wave * 16;

  // Q B-fragments (K-dim = d): loaded once, row-contiguous per lane
  Frag qf[2];
  const __bf16* qp = &Qr[((size_t)bh * S_ + qbase + lrow) * HD_];
  #pragma unroll
  for (int f = 0; f < 2; ++f) {
    int dofs = f * 32 + lhi * 8;
    qf[f].f[0] = *(const f32x4*)(qp + dofs);
    qf[f].f[1] = *(const f32x4*)(qp + dofs + 16);
  }

  // per-thread staging coordinates (2 x 16B async loads per tile)
  const int krow = t >> 3, kch = t & 7;         // K: 32 rows x 8 chunks
  const int vrow = t >> 2, vch = t & 3;         // V^T: 64 rows x 4 chunks

  // prologue: stage tile j=0 into buffer 0
  async_load_b128(&Kr[((size_t)bh * S_ + krow) * HD_ + kch * 8],
                  &ks[0][krow * KPITCH + kch * 8]);
  async_load_b128(&Vt[((size_t)bh * HD_ + vrow) * S_ + vch * 8],
                  &vs[0][vrow * VPITCH + vch * 8]);

  v8f o[4] = {};
  float m = -3.0e38f, l = 0.0f;
  const float* bp = bias + (size_t)b * S_;
  const bool lo = (lhi == 0);
  int cur = 0;

  for (int j = 0; j < S_; j += 32) {
    if (j + 32 < S_) {
      // stream next tile into the other buffer while we compute this one
      async_load_b128(&Kr[((size_t)bh * S_ + j + 32 + krow) * HD_ + kch * 8],
                      &ks[cur ^ 1][krow * KPITCH + kch * 8]);
      async_load_b128(&Vt[((size_t)bh * HD_ + vrow) * S_ + j + 32 + vch * 8],
                      &vs[cur ^ 1][vrow * VPITCH + vch * 8]);
      // async loads complete in order: <=2 outstanding => current tile landed
      asm volatile("s_wait_asynccnt 0x2" ::: "memory");
    } else {
      asm volatile("s_wait_asynccnt 0x0" ::: "memory");
    }
    __syncthreads();                            // current tile visible to all

    const __bf16* ksb = ks[cur];
    const __bf16* vsb = vs[cur];

    // scores^T: T0 = rows j..j+15, T1 = rows j+16..j+31 (q = lane axis)
    v8f T0 = {}, T1 = {};
    #pragma unroll
    for (int kk = 0; kk < 64; kk += 32) {
      Frag ka, kb;
      const __bf16* k0 = &ksb[lrow * KPITCH + kk + lhi * 8];
      ka.f[0] = *(const f32x4*)(k0);
      ka.f[1] = *(const f32x4*)(k0 + 16);
      T0 = wmma_bf16(ka.v, qf[kk >> 5].v, T0);
      const __bf16* k1 = &ksb[(16 + lrow) * KPITCH + kk + lhi * 8];
      kb.f[0] = *(const f32x4*)(k1);
      kb.f[1] = *(const f32x4*)(k1 + 16);
      T1 = wmma_bf16(kb.v, qf[kk >> 5].v, T1);
    }

    // mask bias (per j-row == VGPR axis, contiguous per lane-half)
    F8 bb0, bb1;
    bb0.v[0] = *(const f32x4*)(bp + j + lhi * 8);
    bb0.v[1] = *(const f32x4*)(bp + j + lhi * 8 + 4);
    bb1.v[0] = *(const f32x4*)(bp + j + 16 + lhi * 8);
    bb1.v[1] = *(const f32x4*)(bp + j + 16 + lhi * 8 + 4);

    float t0[8], t1[8], mt = -3.0e38f;
    #pragma unroll
    for (int r = 0; r < 8; ++r) {
      t0[r] = T0[r] + bb0.s[r];
      t1[r] = T1[r] + bb1.s[r];
      mt = fmaxf(mt, fmaxf(t0[r], t1[r]));
    }
    mt = fmaxf(mt, __shfl_xor(mt, 16));         // combine lane halves
    float mnew  = fmaxf(m, mt);
    float alpha = __expf(m - mnew);

    float p0[8], p1[8], rs = 0.0f;
    #pragma unroll
    for (int r = 0; r < 8; ++r) {
      p0[r] = __expf(t0[r] - mnew);
      p1[r] = __expf(t1[r] - mnew);
      rs += p0[r] + p1[r];
    }
    rs += __shfl_xor(rs, 16);
    l = l * alpha + rs;
    m = mnew;
    #pragma unroll
    for (int dt = 0; dt < 4; ++dt)
      #pragma unroll
      for (int r = 0; r < 8; ++r) o[dt][r] *= alpha;

    // build P B-fragment (32(j) x 16(q)) via half swaps
    float p0s[8], p1s[8];
    #pragma unroll
    for (int r = 0; r < 8; ++r) {
      p0s[r] = __shfl_xor(p0[r], 16);
      p1s[r] = __shfl_xor(p1[r], 16);
    }
    Frag pf;
    #pragma unroll
    for (int v = 0; v < 4; ++v) {
      pf.e[2 * v]     = (__bf16)(lo ? p0[2 * v]     : p1s[2 * v]);
      pf.e[2 * v + 1] = (__bf16)(lo ? p0[2 * v + 1] : p1s[2 * v + 1]);
    }
    #pragma unroll
    for (int v = 4; v < 8; ++v) {
      pf.e[2 * v]     = (__bf16)(lo ? p0s[2 * v - 8] : p1[2 * v - 8]);
      pf.e[2 * v + 1] = (__bf16)(lo ? p0s[2 * v - 7] : p1[2 * v - 7]);
    }

    // O^T += V^T(d x j) * P(j x q)
    #pragma unroll
    for (int dt = 0; dt < 4; ++dt) {
      Frag va;
      const __bf16* vsrc = &vsb[(dt * 16 + lrow) * VPITCH + lhi * 16];
      va.f[0] = *(const f32x4*)(vsrc);
      va.f[1] = *(const f32x4*)(vsrc + 8);
      o[dt] = wmma_bf16(va.v, pf.v, o[dt]);
    }

    __syncthreads();   // all reads of buffer `cur` done before it is restaged
    cur ^= 1;
  }

  // normalize + store AO[b, s, head*64 + d] (d contiguous per lane)
  const float linv = 1.0f / l;
  __bf16* aop = &AO[((size_t)(b * S_ + qbase + lrow)) * H_ + head * 64 + lhi * 8];
  #pragma unroll
  for (int dt = 0; dt < 4; ++dt) {
    Frag oo;
    #pragma unroll
    for (int r = 0; r < 8; ++r) oo.e[r] = (__bf16)(o[dt][r] * linv);
    *(f32x4*)(&aop[dt * 16]) = oo.f[0];
  }
}

// ------------------------- output projection -------------------------------

__global__ __launch_bounds__(256) void k_proj(
    const __bf16* __restrict__ AO, const __bf16* __restrict__ Wob,
    float* __restrict__ out) {
  __shared__ __align__(16) __bf16 xs[16 * XPITCH];
  const int mbase = blockIdx.x * 16;
  const int t = threadIdx.x;
  #pragma unroll
  for (int c = 0; c < 4; ++c) {
    int idx = t + c * 256;
    int row = idx >> 6, col = idx & 63;
    *(f32x4*)(&xs[row * XPITCH + col * 8]) =
        *(const f32x4*)(&AO[(size_t)(mbase + row) * H_ + col * 8]);
  }
  __syncthreads();

  const int wave = t >> 5, lane = t & 31;
  const int lrow = lane & 15, lhi = lane >> 4;
  const int chunk = blockIdx.y * 8 + wave;      // 0..15
  const int nbase = chunk * 64;

  v8f acc[4] = {};
  for (int k = 0; k < H_; k += 32) {
    Frag bf;
    const __bf16* xsrc = &xs[lrow * XPITCH + k + lhi * 8];
    bf.f[0] = *(const f32x4*)(xsrc);
    bf.f[1] = *(const f32x4*)(xsrc + 16);
    #pragma unroll
    for (int dt = 0; dt < 4; ++dt) {
      Frag af;
      const __bf16* wsrc = &Wob[(size_t)(nbase + dt * 16 + lrow) * H_ + k + lhi * 8];
      af.f[0] = *(const f32x4*)(wsrc);
      af.f[1] = *(const f32x4*)(wsrc + 16);
      acc[dt] = wmma_bf16(af.v, bf.v, acc[dt]);
    }
  }

  float* dst = &out[(size_t)(mbase + lrow) * H_ + nbase + lhi * 8];
  #pragma unroll
  for (int dt = 0; dt < 4; ++dt) {
    f32x4 a, b;
    a.x = acc[dt][0]; a.y = acc[dt][1]; a.z = acc[dt][2]; a.w = acc[dt][3];
    b.x = acc[dt][4]; b.y = acc[dt][5]; b.z = acc[dt][6]; b.w = acc[dt][7];
    *(f32x4*)(&dst[dt * 16])     = a;
    *(f32x4*)(&dst[dt * 16 + 4]) = b;
  }
}

// ------------------------- host launcher -----------------------------------

extern "C" void kernel_launch(void* const* d_in, const int* in_sizes, int n_in,
                              void* d_out, int out_size, void* d_ws, size_t ws_size,
                              hipStream_t stream) {
  const float* x    = (const float*)d_in[0];
  const int*   mask = (const int*)d_in[1];
  const float* Wq   = (const float*)d_in[2];
  const float* Wk   = (const float*)d_in[3];
  const float* Wv   = (const float*)d_in[4];
  const float* Wo   = (const float*)d_in[5];
  float* out = (float*)d_out;

  char* ws = (char*)d_ws;
  const size_t MB = 1ull << 20;
  __bf16* xb  = (__bf16*)(ws);              // 16MB; reused as AO after k_qkv
  __bf16* AO  = (__bf16*)(ws);
  __bf16* wqb = (__bf16*)(ws + 16 * MB);    // 2MB each
  __bf16* wkb = (__bf16*)(ws + 18 * MB);
  __bf16* wvb = (__bf16*)(ws + 20 * MB);
  __bf16* wob = (__bf16*)(ws + 22 * MB);
  __bf16* Qr  = (__bf16*)(ws + 24 * MB);    // 16MB
  __bf16* Kr  = (__bf16*)(ws + 40 * MB);    // 16MB
  __bf16* Vt  = (__bf16*)(ws + 56 * MB);    // 16MB
  float*  cosT = (float*)(ws + 72 * MB);
  float*  sinT = (float*)(ws + 72 * MB + 256 * 1024);
  float*  bias = (float*)(ws + 72 * MB + 512 * 1024);

  const int NX = B_ * S_ * H_;              // 8M
  const int NW = H_ * H_;                   // 1M

  k_cvt<<<NX / 1024, 256, 0, stream>>>(x,  xb,  NX / 4);
  k_cvt<<<NW / 1024, 256, 0, stream>>>(Wq, wqb, NW / 4);
  k_cvt<<<NW / 1024, 256, 0, stream>>>(Wk, wkb, NW / 4);
  k_cvt<<<NW / 1024, 256, 0, stream>>>(Wv, wvb, NW / 4);
  k_cvt<<<NW / 1024, 256, 0, stream>>>(Wo, wob, NW / 4);
  k_tables<<<(S_ * 32) / 256, 256, 0, stream>>>(cosT, sinT);
  k_bias<<<(B_ * S_) / 256, 256, 0, stream>>>(mask, bias, B_ * S_);

  k_qkv<<<dim3((B_ * S_) / 16, 6), 256, 0, stream>>>(xb, wqb, wkb, wvb,
                                                     cosT, sinT, Qr, Kr, Vt);
  k_attn<<<dim3(B_ * NH_, S_ / 128), 256, 0, stream>>>(Qr, Kr, Vt, bias, AO);
  k_proj<<<dim3((B_ * S_) / 16, 2), 256, 0, stream>>>(AO, wob, out);
}
